// mamba_unit_77996606095863
// MI455X (gfx1250) — matrix operand order repaired and checked by reading, not compiled, not run on hardware
//
#include <hip/hip_runtime.h>
#include <cmath>

typedef _Float16 v16h __attribute__((ext_vector_type(16)));
typedef _Float16 v8h  __attribute__((ext_vector_type(8)));
typedef float    v8f  __attribute__((ext_vector_type(8)));

#define T_LEN   2048
#define B_SZ    16
#define NTOK    (T_LEN * B_SZ)     // 32768
#define DMODEL  512
#define DINNER  1024
#define DSTATE  16
#define DTRANK  32
#define NUNITS  100
#define NUPAD   112
#define W1ROWS  128                // w1 padded to full 64-row tiles (zero rows 100..127)
#define LAT     128

__device__ __forceinline__ float silu_f(float x) { return x / (1.f + __expf(-x)); }

// ---- CDNA5 async global->LDS copy (ASYNCcnt-tracked, bypasses VGPRs) ----
// Generic shared-aperture addresses carry the workgroup-relative LDS byte
// address in their low 32 bits (ISA 10.2), so truncation gives the VDST value.
__device__ __forceinline__ void async_load_b128(const void* gptr, void* lptr) {
    unsigned lds = (unsigned)(size_t)lptr;
    unsigned long long ga = (unsigned long long)(size_t)gptr;
    asm volatile("global_load_async_to_lds_b128 %0, %1, off"
                 :: "v"(lds), "v"(ga) : "memory");
}
__device__ __forceinline__ void wait_async0() {
    asm volatile("s_wait_asynccnt 0x0" ::: "memory");
}

// ---------------- weight conversion ----------------
__global__ void f32_to_f16_kernel(const float* __restrict__ src, _Float16* __restrict__ dst, int n) {
    int i = blockIdx.x * blockDim.x + threadIdx.x;
    if (i < n) dst[i] = (_Float16)src[i];
}

__global__ void pack_w1_kernel(const float* __restrict__ w1, const float* __restrict__ b1,
                               _Float16* __restrict__ w1h, float* __restrict__ b1p) {
    int i = blockIdx.x * blockDim.x + threadIdx.x;      // over W1ROWS*DMODEL
    if (i >= W1ROWS * DMODEL) return;
    int r = i / DMODEL, c = i % DMODEL;
    w1h[i] = (r < NUNITS) ? (_Float16)w1[r * DMODEL + c] : (_Float16)0.f;
    if (c == 0 && r < NUPAD) b1p[r] = (r < NUNITS) ? b1[r] : 0.f;
}

// -------- build u: concat(y_mean,y_std,x), [T,B]->[B,T], f16 --------
__global__ void build_u_kernel(const float* __restrict__ y_mean, const float* __restrict__ y_std,
                               const float* __restrict__ x, _Float16* __restrict__ u_h) {
    int i = blockIdx.x * blockDim.x + threadIdx.x;      // over NTOK*DMODEL
    if (i >= NTOK * DMODEL) return;
    int n = i >> 9, c = i & 511;
    int b = n >> 11, t = n & (T_LEN - 1);
    int tb = t * B_SZ + b;
    float v;
    if (c < LAT)          v = y_mean[(size_t)tb * LAT + c];
    else if (c < 2 * LAT) v = y_std [(size_t)tb * LAT + (c - LAT)];
    else                  v = x     [(size_t)tb * 256 + (c - 2 * LAT)];
    u_h[i] = (_Float16)v;
}

// ---------------- WMMA GEMM: C[M,N] = A[M,K] * W[N,K]^T ----------------
// MODE 0: f16 out; MODE 1: f32 out; MODE 2: f16 out = tanh(acc + bias[col]).
// Workgroup tile 128x64, 8 waves, each wave a 32x32 patch (2x2 WMMA tiles), K step 32.
// Tiles are staged into LDS with global_load_async_to_lds_b128 (ASYNCcnt),
// so W must be readable (and zero where unused) for all rows of each 64-row tile.
template<int MODE>
__global__ void gemm_wmma_f16(const _Float16* __restrict__ A, const _Float16* __restrict__ W,
                              void* __restrict__ C, const float* __restrict__ bias,
                              int M, int N, int K) {
    __shared__ __align__(16) _Float16 As[128 * 40];
    __shared__ __align__(16) _Float16 Ws[64 * 40];
    const int tid  = threadIdx.x;
    const int lane = tid & 31, wave = tid >> 5;
    const int wmo = (wave >> 1) * 32;       // wave row offset within block
    const int wno = (wave & 1) * 32;        // wave col offset within block
    const int bm = blockIdx.y * 128, bn = blockIdx.x * 64;
    const int g = lane >> 4, mm = lane & 15;

    v8f acc[2][2];
    v8f zero = {};
    #pragma unroll
    for (int i = 0; i < 2; ++i)
        #pragma unroll
        for (int j = 0; j < 2; ++j) acc[i][j] = zero;

    const int ar  = tid >> 1, ac  = (tid & 1) * 16;     // A tile: 128 rows x 32 halves
    const int lwr = tid >> 2, lwc = (tid & 3) * 8;      // W tile: 64 rows x 32 halves

    for (int kb = 0; kb < K; kb += 32) {
        const _Float16* asrc = A + (size_t)(bm + ar) * K + kb + ac;
        const _Float16* wsrc = W + (size_t)(bn + lwr) * K + kb + lwc;
        async_load_b128(asrc,     &As[ar * 40 + ac]);
        async_load_b128(asrc + 8, &As[ar * 40 + ac + 8]);
        async_load_b128(wsrc,     &Ws[lwr * 40 + lwc]);
        if (kb + 32 < K) __builtin_prefetch(asrc + 32, 0, 1);  // global_prefetch next K tile
        wait_async0();          // own wave's async copies landed in LDS
        __syncthreads();        // all waves' copies landed

        v16h afr[2], bfr[2];
        // A fragment: lanes 0-15 and 16-31 both carry rows M=0..15; lane group g
        // selects K halves {g*8..g*8+7} (halves 0..7) and {16+g*8..} (halves 8..15).
        #pragma unroll
        for (int i = 0; i < 2; ++i) {
            const _Float16* p = &As[(wmo + i * 16 + mm) * 40];
            union { v16h v; v8h h[2]; } u;
            u.h[0] = *(const v8h*)(p + g * 8);
            u.h[1] = *(const v8h*)(p + 16 + g * 8);
            afr[i] = u.v;
        }
        // B fragment: lane n = lane&15 is the output column; g selects K {g*8..g*8+7} and {16+g*8..16+g*8+7}.
        #pragma unroll
        for (int j = 0; j < 2; ++j) {
            const _Float16* q = &Ws[(wno + j * 16 + mm) * 40 + g * 8];
            union { v16h v; v8h h[2]; } u;
            u.h[0] = *(const v8h*)q;
            u.h[1] = *(const v8h*)(q + 16);
            bfr[j] = u.v;
        }
        #pragma unroll
        for (int i = 0; i < 2; ++i)
            #pragma unroll
            for (int j = 0; j < 2; ++j)
                acc[i][j] = __builtin_amdgcn_wmma_f32_16x16x32_f16(
                    false, afr[i], false, bfr[j], (short)0, acc[i][j], false, false);
        __syncthreads();
    }
    // C/D layout: VGPR v -> row = v + g*8 (lanes 0-15: M=v, lanes 16-31: M=v+8); col = lane&15.
    #pragma unroll
    for (int i = 0; i < 2; ++i) {
        #pragma unroll
        for (int j = 0; j < 2; ++j) {
            int col = bn + wno + j * 16 + mm;
            if (col >= N) continue;
            #pragma unroll
            for (int v = 0; v < 8; ++v) {
                int row = bm + wmo + i * 16 + g * 8 + v;
                float val = acc[i][j][v];
                if (MODE == 1)      ((float*)C)[(size_t)row * N + col] = val;
                else if (MODE == 2) ((_Float16*)C)[(size_t)row * N + col] = (_Float16)tanhf(val + bias[col]);
                else                ((_Float16*)C)[(size_t)row * N + col] = (_Float16)val;
            }
        }
    }
}

// ---------------- depthwise causal conv (d_conv=4) + SiLU ----------------
__global__ void conv_silu_kernel(const _Float16* __restrict__ xz, const float* __restrict__ conv_w,
                                 const float* __restrict__ conv_b, _Float16* __restrict__ xc) {
    int i = blockIdx.x * blockDim.x + threadIdx.x;      // NTOK*DINNER
    if (i >= NTOK * DINNER) return;
    int n = i >> 10, d = i & (DINNER - 1);
    int b = n >> 11, t = n & (T_LEN - 1);
    float acc = conv_b[d];
    #pragma unroll
    for (int k = 0; k < 4; ++k) {
        int tt = t - 3 + k;
        if (tt >= 0)
            acc += (float)xz[(size_t)(b * T_LEN + tt) * (2 * DINNER) + d] * conv_w[d * 4 + k];
    }
    xc[i] = (_Float16)silu_f(acc);
}

// ---------------- dt = softplus(x_dbl[:, :32] @ dt_proj_w^T + b) ----------------
__global__ void dt_kernel(const float* __restrict__ x_dbl, const float* __restrict__ dtw,
                          const float* __restrict__ dtb, float* __restrict__ dt) {
    int i = blockIdx.x * blockDim.x + threadIdx.x;      // NTOK*DINNER
    if (i >= NTOK * DINNER) return;
    int n = i >> 10, d = i & (DINNER - 1);
    const float* xr = x_dbl + (size_t)n * 64;
    const float* wr = dtw + d * DTRANK;
    float acc = dtb[d];
    #pragma unroll
    for (int r = 0; r < DTRANK; ++r) acc += xr[r] * wr[r];
    dt[i] = acc > 20.f ? acc : log1pf(__expf(acc));
}

// ---------------- selective scan: h in registers, B/C staged in LDS ----------------
__global__ void scan_kernel(const _Float16* __restrict__ xz, const _Float16* __restrict__ xc,
                            const float* __restrict__ dt, const float* __restrict__ x_dbl,
                            const float* __restrict__ A_log, const float* __restrict__ Dv,
                            _Float16* __restrict__ yg) {
    const int d = blockIdx.x * 128 + threadIdx.x;
    const int b = blockIdx.y;
    float Arow[DSTATE];
    #pragma unroll
    for (int s = 0; s < DSTATE; ++s) Arow[s] = -__expf(A_log[d * DSTATE + s]);
    const float Dd = Dv[d];
    float h[DSTATE];
    #pragma unroll
    for (int s = 0; s < DSTATE; ++s) h[s] = 0.f;
    __shared__ __align__(16) float bc[2][32];           // B[16] | C[16], double-buffered
    for (int t = 0; t < T_LEN; ++t) {
        const int n = (b << 11) + t;
        if (threadIdx.x < 8)                             // 8 lanes x 16B = 128B of B|C
            async_load_b128(x_dbl + (size_t)n * 64 + 32 + threadIdx.x * 4,
                            &bc[t & 1][threadIdx.x * 4]);
        wait_async0();
        __syncthreads();
        float dtv = dt[(size_t)n * DINNER + d];
        float xv  = (float)xc[(size_t)n * DINNER + d];
        float dx  = dtv * xv;
        float y   = 0.f;
        const float* Bc = bc[t & 1];
        const float* Cc = bc[t & 1] + 16;
        #pragma unroll
        for (int s = 0; s < DSTATE; ++s) {
            float e = __expf(dtv * Arow[s]);
            h[s] = e * h[s] + dx * Bc[s];
            y += h[s] * Cc[s];
        }
        float zv = (float)xz[(size_t)n * (2 * DINNER) + DINNER + d];
        y = (y + Dd * xv) * silu_f(zv);
        yg[(size_t)n * DINNER + d] = (_Float16)y;
    }
}

// ---------------- head: out = h1 @ w2^T + b2, then mask blend ----------------
__global__ void head_kernel(const _Float16* __restrict__ h1, const float* __restrict__ w2,
                            const float* __restrict__ b2, const float* __restrict__ x,
                            const float* __restrict__ y_mean, const float* __restrict__ y_std,
                            float* __restrict__ out) {
    const int n = blockIdx.x;                           // token in [B,T] order
    const int b = n >> 11, t = n & (T_LEN - 1);
    const int tb = t * B_SZ + b;
    __shared__ float hb[NUNITS];
    __shared__ float mb[128];
    const int tid = threadIdx.x;
    if (tid < NUNITS) hb[tid] = (float)h1[(size_t)n * NUPAD + tid];
    if (tid < 128)    mb[tid] = x[(size_t)tb * 256 + 128 + tid];
    __syncthreads();
    float ms = 0.f;
    for (int j = 0; j < 128; ++j) ms += mb[j];
    const bool m = ms > 0.f;
    const int o = tid;                                  // 0..255
    float acc = b2[o];
    const float* wr = w2 + o * NUNITS;
    for (int j = 0; j < NUNITS; ++j) acc += hb[j] * wr[j];
    if (o < LAT) {
        out[(size_t)tb * LAT + o] = m ? acc : y_mean[(size_t)tb * LAT + o];
    } else {
        int oo = o - LAT;
        float res = m ? acc : y_std[(size_t)tb * LAT + oo];
        out[(size_t)NTOK * LAT + (size_t)tb * LAT + oo] = fabsf(res);
    }
}

extern "C" void kernel_launch(void* const* d_in, const int* in_sizes, int n_in,
                              void* d_out, int out_size, void* d_ws, size_t ws_size,
                              hipStream_t stream) {
    const float* y_mean     = (const float*)d_in[0];
    const float* y_std      = (const float*)d_in[1];
    const float* x          = (const float*)d_in[2];
    const float* in_proj_w  = (const float*)d_in[3];
    const float* conv_w     = (const float*)d_in[4];
    const float* conv_b     = (const float*)d_in[5];
    const float* x_proj_w   = (const float*)d_in[6];
    const float* dt_proj_w  = (const float*)d_in[7];
    const float* dt_proj_b  = (const float*)d_in[8];
    const float* A_log      = (const float*)d_in[9];
    const float* Dv         = (const float*)d_in[10];
    const float* out_proj_w = (const float*)d_in[11];
    const float* w1         = (const float*)d_in[12];
    const float* b1         = (const float*)d_in[13];
    const float* w2         = (const float*)d_in[14];
    const float* b2         = (const float*)d_in[15];
    (void)in_sizes; (void)n_in; (void)out_size; (void)ws_size;

    char* base = (char*)d_ws;
    size_t off = 0;
    auto carve = [&](size_t bytes) -> char* {
        char* p = base + off;
        off += (bytes + 255) & ~(size_t)255;
        return p;
    };
    _Float16* u_h   = (_Float16*)carve((size_t)NTOK * DMODEL * 2);
    _Float16* inp16 = (_Float16*)carve((size_t)2 * DINNER * DMODEL * 2);
    _Float16* xp16  = (_Float16*)carve((size_t)64 * DINNER * 2);
    _Float16* op16  = (_Float16*)carve((size_t)DMODEL * DINNER * 2);
    _Float16* w1h   = (_Float16*)carve((size_t)W1ROWS * DMODEL * 2);
    float*    b1p   = (float*)carve((size_t)NUPAD * 4);
    _Float16* xz16  = (_Float16*)carve((size_t)NTOK * 2 * DINNER * 2);
    _Float16* xc16  = (_Float16*)carve((size_t)NTOK * DINNER * 2);
    float*    xdbl  = (float*)carve((size_t)NTOK * 64 * 4);
    float*    dt32  = (float*)carve((size_t)NTOK * DINNER * 4);
    _Float16* yg16  = (_Float16*)carve((size_t)NTOK * DINNER * 2);
    _Float16* yc16  = (_Float16*)carve((size_t)NTOK * DMODEL * 2);
    _Float16* h1h   = (_Float16*)carve((size_t)NTOK * NUPAD * 2);

    f32_to_f16_kernel<<<(2 * DINNER * DMODEL + 255) / 256, 256, 0, stream>>>(in_proj_w, inp16, 2 * DINNER * DMODEL);
    f32_to_f16_kernel<<<(64 * DINNER + 255) / 256, 256, 0, stream>>>(x_proj_w, xp16, 64 * DINNER);
    f32_to_f16_kernel<<<(DMODEL * DINNER + 255) / 256, 256, 0, stream>>>(out_proj_w, op16, DMODEL * DINNER);
    pack_w1_kernel<<<(W1ROWS * DMODEL + 255) / 256, 256, 0, stream>>>(w1, b1, w1h, b1p);
    build_u_kernel<<<(NTOK * DMODEL + 255) / 256, 256, 0, stream>>>(y_mean, y_std, x, u_h);

    // in_proj: [32768,512] x [2048,512]^T -> xz f16
    gemm_wmma_f16<0><<<dim3((2 * DINNER) / 64, NTOK / 128), 256, 0, stream>>>(
        u_h, inp16, xz16, nullptr, NTOK, 2 * DINNER, DMODEL);
    conv_silu_kernel<<<(NTOK * DINNER + 255) / 256, 256, 0, stream>>>(xz16, conv_w, conv_b, xc16);
    // x_proj: [32768,1024] x [64,1024]^T -> x_dbl f32
    gemm_wmma_f16<1><<<dim3(1, NTOK / 128), 256, 0, stream>>>(
        xc16, xp16, xdbl, nullptr, NTOK, 64, DINNER);
    dt_kernel<<<(NTOK * DINNER + 255) / 256, 256, 0, stream>>>(xdbl, dt_proj_w, dt_proj_b, dt32);
    scan_kernel<<<dim3(DINNER / 128, B_SZ), 128, 0, stream>>>(xz16, xc16, dt32, xdbl, A_log, Dv, yg16);
    // out_proj: [32768,1024] x [512,1024]^T -> yc f16
    gemm_wmma_f16<0><<<dim3(DMODEL / 64, NTOK / 128), 256, 0, stream>>>(
        yg16, op16, yc16, nullptr, NTOK, DMODEL, DINNER);
    // MLP hidden: [32768,512] x [128-pad,512]^T, tanh(acc+bias) epilogue, store 112 cols
    gemm_wmma_f16<2><<<dim3((NUPAD + 63) / 64, NTOK / 128), 256, 0, stream>>>(
        yc16, w1h, h1h, b1p, NTOK, NUPAD, DMODEL);
    head_kernel<<<NTOK, 256, 0, stream>>>(h1h, w2, b2, x, y_mean, y_std, (float*)d_out);
}
